// MultiHeadAttention_13537736917853
// MI455X (gfx1250) — matrix operand order repaired and checked by reading, not compiled, not run on hardware
//
#include <hip/hip_runtime.h>

// Problem constants (from reference)
#define B_  8
#define L_  1024
#define D_  1024
#define H_  16
#define HD_ 64

typedef __attribute__((ext_vector_type(16))) _Float16 v16h;
typedef __attribute__((ext_vector_type(8)))  _Float16 v8h;
typedef __attribute__((ext_vector_type(4)))  _Float16 v4h;
typedef __attribute__((ext_vector_type(8)))  float    v8f;

// ---------------------------------------------------------------------------
// Fragment loaders (wave32, v_wmma_f32_16x16x32_f16 operands, contiguous k = 16*(lane/16)+s in both A and B)
// ---------------------------------------------------------------------------

// A-matrix 16x32 fragment from an f16 row. p = &A[row][kb], row = tile_m+(lane&15).
// Lanes 0-15: K = kb..kb+15; lanes 16-31: K = kb+16..kb+31.
__device__ __forceinline__ v16h load_a_f16(const _Float16* __restrict__ p, int lane) {
  const int sel = (lane & 16) ? 16 : 0;
  v8h lo = *reinterpret_cast<const v8h*>(p + sel);
  v8h hi = *reinterpret_cast<const v8h*>(p + 8 + sel);
  v16h f;
#pragma unroll
  for (int i = 0; i < 8; ++i) { f[i] = lo[i]; f[8 + i] = hi[i]; }
  return f;
}

// B-matrix 32x16 fragment where the lane's column is a contiguous f16 row
// (covers both W^T operands and rows of per-head-transposed V).
// Lanes 0-15 hold K = kb..kb+15, lanes 16-31 hold K = kb+16..kb+31.
__device__ __forceinline__ v16h load_b_f16(const _Float16* __restrict__ p, int lane) {
  const int koff = (lane & 16) ? 16 : 0;
  return *reinterpret_cast<const v16h*>(p + koff);
}

// A-matrix fragment with on-the-fly fp32 -> f16 conversion (used only for the
// attention-probability operand, which must live in fp32 as a kernel output).
__device__ __forceinline__ v16h load_a_f32cvt(const float* __restrict__ p, int lane) {
  const int sel = (lane & 16) ? 16 : 0;
  const float4* q0 = reinterpret_cast<const float4*>(p + sel);
  const float4* q1 = reinterpret_cast<const float4*>(p + 8 + sel);
  float4 x0 = q0[0], x1 = q0[1], y0 = q1[0], y1 = q1[1];
  v16h f;
  f[0]  = (_Float16)x0.x; f[1]  = (_Float16)x0.y; f[2]  = (_Float16)x0.z; f[3]  = (_Float16)x0.w;
  f[4]  = (_Float16)x1.x; f[5]  = (_Float16)x1.y; f[6]  = (_Float16)x1.z; f[7]  = (_Float16)x1.w;
  f[8]  = (_Float16)y0.x; f[9]  = (_Float16)y0.y; f[10] = (_Float16)y0.z; f[11] = (_Float16)y0.w;
  f[12] = (_Float16)y1.x; f[13] = (_Float16)y1.y; f[14] = (_Float16)y1.z; f[15] = (_Float16)y1.w;
  return f;
}

__device__ __forceinline__ v8f wmma_f16(v16h a, v16h b, v8f c) {
  return __builtin_amdgcn_wmma_f32_16x16x32_f16(false, a, false, b, (short)0, c, false, false);
}

// ---------------------------------------------------------------------------
// fp32 -> f16 conversion pass (bandwidth bound; done ONCE so GEMM loops are
// conversion-free). Each thread converts 4 elements.
// ---------------------------------------------------------------------------
__global__ __launch_bounds__(256) void cvt_f32_to_f16(
    const float* __restrict__ src, _Float16* __restrict__ dst) {
  const size_t i = (size_t)blockIdx.x * blockDim.x + threadIdx.x;
  float4 x = reinterpret_cast<const float4*>(src)[i];
  v4h h;
  h[0] = (_Float16)x.x; h[1] = (_Float16)x.y;
  h[2] = (_Float16)x.z; h[3] = (_Float16)x.w;
  reinterpret_cast<v4h*>(dst)[i] = h;
}

// ---------------------------------------------------------------------------
// C = A[M,K] @ W[N,K]^T + bias[N], all-f16 operands, f32 accumulate.
// Block: 256 threads = 8 waves (2x4), block tile 64(M) x 256(N),
// wave tile 32x64 -> 8 WMMAs per 32-wide K step, 12 b128 loads, 0 converts.
// MODE 0: C fp32 [M,N].  MODE 1: C f16 [M,N].
// MODE 2: C f16, written per-head transposed: Vt[((b*H+h)*HD+nh)*L + l].
// ---------------------------------------------------------------------------
template <int MODE>
__global__ __launch_bounds__(256) void gemm_f16(
    const _Float16* __restrict__ A, const _Float16* __restrict__ W,
    const float* __restrict__ bias, void* __restrict__ Cout,
    int M, int N, int K) {
  const int lane = threadIdx.x & 31;
  const int wave = threadIdx.x >> 5;
  const int m0 = blockIdx.x * 64  + (wave >> 2) * 32;
  const int n0 = blockIdx.y * 256 + (wave & 3) * 64;
  const int col = lane & 15;

  v8f acc[2][4] = {};

  const _Float16* Ar0 = A + (size_t)(m0 + col) * K;
  const _Float16* Ar1 = A + (size_t)(m0 + 16 + col) * K;
  const _Float16* Wr[4];
#pragma unroll
  for (int j = 0; j < 4; ++j) Wr[j] = W + (size_t)(n0 + 16 * j + col) * K;

  for (int kb = 0; kb < K; kb += 32) {
    if (kb + 32 < K) {  // stream-ahead: lowers to global_prefetch_b8
      __builtin_prefetch(Ar0 + kb + 32, 0, 1);
      __builtin_prefetch(Ar1 + kb + 32, 0, 1);
    }
    v16h a0 = load_a_f16(Ar0 + kb, lane);
    v16h a1 = load_a_f16(Ar1 + kb, lane);
#pragma unroll
    for (int j = 0; j < 4; ++j) {
      v16h b = load_b_f16(Wr[j] + kb, lane);
      acc[0][j] = wmma_f16(a0, b, acc[0][j]);
      acc[1][j] = wmma_f16(a1, b, acc[1][j]);
    }
  }

  const int ro = (lane & 16) ? 8 : 0;
#pragma unroll
  for (int i = 0; i < 2; ++i)
#pragma unroll
    for (int j = 0; j < 4; ++j) {
      const int n = n0 + 16 * j + col;
      const float bv = bias[n];
#pragma unroll
      for (int r = 0; r < 8; ++r) {
        const int row = m0 + 16 * i + ro + r;
        const float val = acc[i][j][r] + bv;
        if (MODE == 0) {
          ((float*)Cout)[(size_t)row * N + n] = val;
        } else if (MODE == 1) {
          ((_Float16*)Cout)[(size_t)row * N + n] = (_Float16)val;
        } else {  // per-head transpose: row = b*L + l, n = h*HD + nh
          const int b = row >> 10, l = row & (L_ - 1);
          const int h = n >> 6, nh = n & (HD_ - 1);
          ((_Float16*)Cout)[(((size_t)b * H_ + h) * HD_ + nh) * L_ + l] = (_Float16)val;
        }
      }
    }
}

// ---------------------------------------------------------------------------
// scores[z, q, k] = scale * sum_d Qh[q,d] * Kh[k,d]   (z = b*H + h)
// Pure f16 loads, K-dim = HD = 64 (2 unrolled WMMA steps).
// ---------------------------------------------------------------------------
__global__ __launch_bounds__(256) void attn_scores(
    const _Float16* __restrict__ Q, const _Float16* __restrict__ Km,
    float* __restrict__ attn, float scale) {
  const int lane = threadIdx.x & 31;
  const int wave = threadIdx.x >> 5;
  const int m0 = blockIdx.x * 64  + (wave >> 2) * 32;
  const int n0 = blockIdx.y * 256 + (wave & 3) * 64;
  const int z = blockIdx.z;
  const int b = z >> 4, h = z & 15;
  const int col = lane & 15;

  const _Float16* Qb = Q  + (size_t)b * L_ * D_ + h * HD_;
  const _Float16* Kb = Km + (size_t)b * L_ * D_ + h * HD_;

  v8f acc[2][4] = {};

  const _Float16* Qr0 = Qb + (size_t)(m0 + col) * D_;
  const _Float16* Qr1 = Qb + (size_t)(m0 + 16 + col) * D_;
  const _Float16* Kr[4];
#pragma unroll
  for (int j = 0; j < 4; ++j) Kr[j] = Kb + (size_t)(n0 + 16 * j + col) * D_;

#pragma unroll
  for (int kb = 0; kb < HD_; kb += 32) {
    v16h a0 = load_a_f16(Qr0 + kb, lane);
    v16h a1 = load_a_f16(Qr1 + kb, lane);
#pragma unroll
    for (int j = 0; j < 4; ++j) {
      v16h bfr = load_b_f16(Kr[j] + kb, lane);
      acc[0][j] = wmma_f16(a0, bfr, acc[0][j]);
      acc[1][j] = wmma_f16(a1, bfr, acc[1][j]);
    }
  }

  float* S = attn + (size_t)z * L_ * L_;
  const int ro = (lane & 16) ? 8 : 0;
#pragma unroll
  for (int i = 0; i < 2; ++i)
#pragma unroll
    for (int j = 0; j < 4; ++j) {
      const int n = n0 + 16 * j + col;
#pragma unroll
      for (int r = 0; r < 8; ++r)
        S[(size_t)(m0 + 16 * i + ro + r) * L_ + n] = scale * acc[i][j][r];
    }
}

// ---------------------------------------------------------------------------
// In-place row softmax over attn rows of length L_ (one block per row).
// ---------------------------------------------------------------------------
__global__ __launch_bounds__(256) void softmax_rows(float* __restrict__ attn) {
  float* p = attn + (size_t)blockIdx.x * L_;
  __shared__ float red[256];
  const int t = threadIdx.x;

  float x0 = p[t], x1 = p[t + 256], x2 = p[t + 512], x3 = p[t + 768];
  float m = fmaxf(fmaxf(x0, x1), fmaxf(x2, x3));
  red[t] = m; __syncthreads();
  for (int s = 128; s > 0; s >>= 1) {
    if (t < s) red[t] = fmaxf(red[t], red[t + s]);
    __syncthreads();
  }
  m = red[0]; __syncthreads();

  x0 = __expf(x0 - m); x1 = __expf(x1 - m);
  x2 = __expf(x2 - m); x3 = __expf(x3 - m);
  red[t] = x0 + x1 + x2 + x3; __syncthreads();
  for (int s = 128; s > 0; s >>= 1) {
    if (t < s) red[t] += red[t + s];
    __syncthreads();
  }
  const float inv = 1.0f / red[0];

  p[t] = x0 * inv; p[t + 256] = x1 * inv;
  p[t + 512] = x2 * inv; p[t + 768] = x3 * inv;
}

// ---------------------------------------------------------------------------
// OH[b, q, h*HD+n] = sum_k P[z,q,k] * V_h[k,n]   (z = b*H + h)
// A = attn fp32 (converted on the fly, the only in-loop converts left),
// B = Vt f16 rows (contiguous). Wave tile 32(M) x 64(= all of HD).
// ---------------------------------------------------------------------------
__global__ __launch_bounds__(256) void attn_pv(
    const float* __restrict__ attn, const _Float16* __restrict__ Vt,
    _Float16* __restrict__ OH) {
  const int lane = threadIdx.x & 31;
  const int wave = threadIdx.x >> 5;
  const int m0 = blockIdx.x * 256 + wave * 32;
  const int z = blockIdx.z;
  const int b = z >> 4, h = z & 15;
  const int col = lane & 15;

  const float* P = attn + (size_t)z * L_ * L_;
  const _Float16* Vb = Vt + (size_t)z * HD_ * L_;  // Vt[z][n][k], row-contiguous in k

  v8f acc[2][4] = {};

  const float* Pr0 = P + (size_t)(m0 + col) * L_;
  const float* Pr1 = P + (size_t)(m0 + 16 + col) * L_;
  const _Float16* Vr[4];
#pragma unroll
  for (int j = 0; j < 4; ++j) Vr[j] = Vb + (size_t)(16 * j + col) * L_;

  for (int kb = 0; kb < L_; kb += 32) {
    if (kb + 32 < L_) {
      __builtin_prefetch(Pr0 + kb + 32, 0, 1);
      __builtin_prefetch(Pr1 + kb + 32, 0, 1);
    }
    v16h a0 = load_a_f32cvt(Pr0 + kb, lane);
    v16h a1 = load_a_f32cvt(Pr1 + kb, lane);
#pragma unroll
    for (int j = 0; j < 4; ++j) {
      v16h bfr = load_b_f16(Vr[j] + kb, lane);
      acc[0][j] = wmma_f16(a0, bfr, acc[0][j]);
      acc[1][j] = wmma_f16(a1, bfr, acc[1][j]);
    }
  }

  const int ro = (lane & 16) ? 8 : 0;
#pragma unroll
  for (int i = 0; i < 2; ++i)
#pragma unroll
    for (int j = 0; j < 4; ++j) {
#pragma unroll
      for (int r = 0; r < 8; ++r)
        OH[((size_t)b * L_ + m0 + 16 * i + ro + r) * D_ + h * HD_ + 16 * j + col] =
            (_Float16)acc[i][j][r];
    }
}

// ---------------------------------------------------------------------------
extern "C" void kernel_launch(void* const* d_in, const int* in_sizes, int n_in,
                              void* d_out, int out_size, void* d_ws, size_t ws_size,
                              hipStream_t stream) {
  (void)in_sizes; (void)n_in; (void)out_size; (void)ws_size;

  const float* query = (const float*)d_in[0];
  const float* key_  = (const float*)d_in[1];
  const float* value = (const float*)d_in[2];
  const float* Wq = (const float*)d_in[3];
  const float* bq = (const float*)d_in[4];
  const float* Wk = (const float*)d_in[5];
  const float* bk = (const float*)d_in[6];
  const float* Wv = (const float*)d_in[7];
  const float* bv = (const float*)d_in[8];
  const float* Wo = (const float*)d_in[9];
  const float* bo = (const float*)d_in[10];

  float* out  = (float*)d_out;
  float* attn = out + (size_t)B_ * L_ * D_;     // outputs concatenated: out, attn

  const size_t SZ  = (size_t)B_ * L_ * D_;      // 8M elements (activations)
  const size_t WSZ = (size_t)D_ * D_;           // 1M elements (weights)
  _Float16* qa  = (_Float16*)d_ws;              // f16 workspace, ~120 MB total
  _Float16* ka  = qa + SZ;
  _Float16* va  = ka + SZ;
  _Float16* Qh  = va + SZ;
  _Float16* Kh  = Qh + SZ;
  _Float16* Vt  = Kh + SZ;                      // per-head transposed V
  _Float16* OH  = Vt + SZ;
  _Float16* Wqh = OH + SZ;
  _Float16* Wkh = Wqh + WSZ;
  _Float16* Wvh = Wkh + WSZ;
  _Float16* Woh = Wvh + WSZ;

  const int M = B_ * L_;                        // 8192
  dim3 blk(256);

  // One-time fp32 -> f16 conversions (4 elements/thread).
  cvt_f32_to_f16<<<dim3((unsigned)(SZ / 1024)),  blk, 0, stream>>>(query, qa);
  cvt_f32_to_f16<<<dim3((unsigned)(SZ / 1024)),  blk, 0, stream>>>(key_,  ka);
  cvt_f32_to_f16<<<dim3((unsigned)(SZ / 1024)),  blk, 0, stream>>>(value, va);
  cvt_f32_to_f16<<<dim3((unsigned)(WSZ / 1024)), blk, 0, stream>>>(Wq, Wqh);
  cvt_f32_to_f16<<<dim3((unsigned)(WSZ / 1024)), blk, 0, stream>>>(Wk, Wkh);
  cvt_f32_to_f16<<<dim3((unsigned)(WSZ / 1024)), blk, 0, stream>>>(Wv, Wvh);
  cvt_f32_to_f16<<<dim3((unsigned)(WSZ / 1024)), blk, 0, stream>>>(Wo, Woh);

  dim3 gGemm(M / 64, D_ / 256);                 // (128, 4)
  gemm_f16<1><<<gGemm, blk, 0, stream>>>(qa, Wqh, bq, Qh, M, D_, D_);
  gemm_f16<1><<<gGemm, blk, 0, stream>>>(ka, Wkh, bk, Kh, M, D_, D_);
  gemm_f16<2><<<gGemm, blk, 0, stream>>>(va, Wvh, bv, Vt, M, D_, D_);

  dim3 gScore(L_ / 64, L_ / 256, B_ * H_);      // (16, 4, 128)
  attn_scores<<<gScore, blk, 0, stream>>>(Qh, Kh, attn, 0.125f /* 64^-0.5 */);

  softmax_rows<<<dim3((unsigned)(B_ * H_ * L_)), blk, 0, stream>>>(attn);

  dim3 gPV(L_ / 256, 1, B_ * H_);               // (4, 1, 128)
  attn_pv<<<gPV, blk, 0, stream>>>(attn, Vt, OH);

  gemm_f16<0><<<gGemm, blk, 0, stream>>>(OH, Woh, bo, out, M, D_, D_);
}
